// TripletLoss_42279658062624
// MI455X (gfx1250) — matrix operand order repaired and checked by reading, not compiled, not run on hardware
//
#include <hip/hip_runtime.h>
#include <hip/hip_bf16.h>

// ---------------------------------------------------------------------------
// Triplet loss with batch-hard mining, fused:
//   1) normalize rows -> f16 (workspace)
//   2) tiled Gram via v_wmma_f32_16x16x32_f16, fused hardest-pos/neg mining
//      on SQUARED distances (sqrt deferred to per-row finalization).
//      64 rows per block, 2 waves per 16-row M-tile; all waves walk the same
//      column-tile stream (parity split) so 4KB B tiles get ~4-way L0 reuse,
//      cutting L2 traffic ~4x vs one M-tile per block.
//   3) deterministic mean reduction
// ---------------------------------------------------------------------------

typedef _Float16 v4h  __attribute__((ext_vector_type(4)));
typedef _Float16 v8h  __attribute__((ext_vector_type(8)));
typedef _Float16 v16h __attribute__((ext_vector_type(16)));
typedef float    v8f  __attribute__((ext_vector_type(8)));

#define N_ROWS 8192
#define DIM    128
#define MARGIN 0.5f
#define NTILES (N_ROWS / 16)   // 512 column tiles

// ---------------------------------------------------------------------------
// Kernel 1: L2-normalize each row and store as f16, row-major [8192][128].
// One wave (32 lanes) per row; each lane owns 4 consecutive floats.
// ---------------------------------------------------------------------------
__global__ void tl_normalize_f16(const float* __restrict__ emb,
                                 _Float16* __restrict__ ef) {
    const int tid  = threadIdx.x;
    const int lane = tid & 31;
    const int wave = tid >> 5;
    const int row  = blockIdx.x * 8 + wave;   // 1024 blocks * 8 waves = 8192 rows

    const float* p = emb + (size_t)row * DIM + lane * 4;
    float x0 = p[0], x1 = p[1], x2 = p[2], x3 = p[3];
    float ss = x0 * x0 + x1 * x1 + x2 * x2 + x3 * x3;
#pragma unroll
    for (int m = 16; m >= 1; m >>= 1)
        ss += __shfl_xor(ss, m, 32);
    const float scale = 1.0f / fmaxf(__builtin_sqrtf(ss), 1e-12f);

    v4h out;
    out.x = (_Float16)(x0 * scale);
    out.y = (_Float16)(x1 * scale);
    out.z = (_Float16)(x2 * scale);
    out.w = (_Float16)(x3 * scale);
    *(v4h*)(ef + (size_t)row * DIM + lane * 4) = out;
}

// ---------------------------------------------------------------------------
// Helper: assemble a 16-half operand from two contiguous 16-byte loads.
// ---------------------------------------------------------------------------
__device__ __forceinline__ v16h tl_join16(const _Float16* lo_p,
                                          const _Float16* hi_p) {
    v8h lo = *(const v8h*)lo_p;
    v8h hi = *(const v8h*)hi_p;
    return __builtin_shufflevector(lo, hi,
            0, 1, 2, 3, 4, 5, 6, 7, 8, 9, 10, 11, 12, 13, 14, 15);
}

// ---------------------------------------------------------------------------
// Kernel 2: 128 blocks x 8 waves. Block covers 64 rows (4 M-tiles).
// Wave w: M-tile = w>>1, column-tile parity = w&1 -> each 4KB column tile is
// read by 4 waves close together in time (served from L0/WGP$ after the
// first miss). Mining runs on squared distances; diagonal exclusion is a
// scalar per-tile test + loop-invariant lane pattern.
//
// A layout (16-bit A 16x32): lane L holds row M = L&15;
//   lanes 0-15 hold K = 0..15 contiguously, lanes 16-31 hold K = 16..31.
// B layout (16-bit B 32x16): lane L holds column N = L&15;
//   lanes 0-15 hold K = 0..15 contiguously, lanes 16-31 hold K = 16..31.
// C layout: VGPR v, lanes 0-15 -> (M=v, N=lane); lanes 16-31 -> (M=v+8).
// ---------------------------------------------------------------------------
__global__ void tl_gram_mine(const _Float16* __restrict__ ef,
                             const int* __restrict__ labels,
                             float* __restrict__ rowloss) {
    const int tid    = threadIdx.x;
    const int lane   = tid & 31;
    const int wave   = tid >> 5;          // 0..7
    const int mt     = wave >> 1;         // 0..3 : M-tile within block
    const int par    = wave & 1;          // 0/1  : column-tile parity
    const int m0     = blockIdx.x * 64 + mt * 16;
    const bool hiHalf = (lane >= 16);
    const int half8   = hiHalf ? 8 : 0;
    const int lane16  = lane & 15;
    const int diagTileIdx = m0 >> 4;      // the one tile that contains diagonal

    // ---- load A tile (rows m0..m0+15, K = 0..127) once into registers ----
    const _Float16* arow = ef + (size_t)(m0 + lane16) * DIM;
    const int kha = hiHalf ? 16 : 0;      // A half-offset in halves
    v16h a[4];
#pragma unroll
    for (int c = 0; c < 4; ++c)
        a[c] = tl_join16(arow + c * 32 + kha, arow + c * 32 + kha + 8);

    // labels of the 8 rows this lane's accumulator slots correspond to,
    // and the loop-invariant diagonal lane pattern
    int  rowLab[8];
    bool dLane[8];
#pragma unroll
    for (int v = 0; v < 8; ++v) {
        rowLab[v] = labels[m0 + v + half8];
        dLane[v]  = (lane16 == v + half8);
    }

    // running max of masked d^2 (positives) / min of d^2 (negatives)
    float hp2[8], hn2[8];
#pragma unroll
    for (int v = 0; v < 8; ++v) { hp2[v] = 0.0f; hn2[v] = 1.0e30f; }

    const int kb = hiHalf ? 16 : 0;       // B half-offset in halves

    for (int it = 0; it < NTILES / 2; ++it) {
        const int jt     = par + 2 * it;
        const int colRow = jt * 16 + lane16;
        const int colLab = labels[colRow];
        const _Float16* brow = ef + (size_t)colRow * DIM;
        const bool diagT = (jt == diagTileIdx);   // scalar per-wave test

        // prefetch this wave's next tile (jt+2) B row
        if (it + 1 < NTILES / 2) {
            const char* nb = (const char*)(brow + 32 * DIM);
            __builtin_prefetch(nb, 0, 3);
            __builtin_prefetch(nb + 128, 0, 3);
        }

        // issue ALL B loads for this tile before any WMMA
        v16h b[4];
#pragma unroll
        for (int c = 0; c < 4; ++c)
            b[c] = tl_join16(brow + c * 32 + kb, brow + c * 32 + kb + 8);

        v8f acc = {};
#pragma unroll
        for (int c = 0; c < 4; ++c)
            acc = __builtin_amdgcn_wmma_f32_16x16x32_f16(
                    /*neg_a=*/false, a[c], /*neg_b=*/false, b[c],
                    /*c_mod=*/(short)0, acc,
                    /*reuse_a=*/false, /*reuse_b=*/false);

        // epilogue on squared distances (no sqrt in the hot loop)
#pragma unroll
        for (int v = 0; v < 8; ++v) {
            const float d2   = fmaxf(__builtin_fmaf(-2.0f, acc[v], 2.0f), 0.0f);
            const bool  same = (rowLab[v] == colLab);
            const bool  excl = diagT && dLane[v];
            hp2[v] = fmaxf(hp2[v], (same && !excl) ? d2 : 0.0f);
            hn2[v] = fminf(hn2[v], same ? 1.0e30f : d2);
        }

        // loose pacing so the 8 waves stay within L0 reach of each other
        if ((it & 31) == 31) __syncthreads();
    }

    // ---- reduce across the 16 lanes of each half (N dimension) ----
#pragma unroll
    for (int v = 0; v < 8; ++v) {
#pragma unroll
        for (int m = 8; m >= 1; m >>= 1) {
            hp2[v] = fmaxf(hp2[v], __shfl_xor(hp2[v], m, 32));
            hn2[v] = fminf(hn2[v], __shfl_xor(hn2[v], m, 32));
        }
    }

    // ---- merge the two parity waves of each M-tile through LDS ----
    __shared__ float hpL[8][16];
    __shared__ float hnL[8][16];
    if (lane16 == 0) {
#pragma unroll
        for (int v = 0; v < 8; ++v) {
            hpL[wave][half8 + v] = hp2[v];
            hnL[wave][half8 + v] = hn2[v];
        }
    }
    __syncthreads();

    if (tid < 64) {                        // one thread per row in the block
        const int t  = tid >> 4;           // M-tile 0..3
        const int rr = tid & 15;           // row within tile
        const float HP2 = fmaxf(hpL[2 * t][rr], hpL[2 * t + 1][rr]);
        const float HN2 = fminf(hnL[2 * t][rr], hnL[2 * t + 1][rr]);
        // sqrt only once per row (monotone, so order of max/min unchanged)
        const float hp = __builtin_sqrtf(HP2);
        const float hn = __builtin_sqrtf(HN2);
        rowloss[blockIdx.x * 64 + tid] = fmaxf(hp - hn + MARGIN, 0.0f);
    }
}

// ---------------------------------------------------------------------------
// Kernel 3: deterministic mean over 8192 row losses.
// ---------------------------------------------------------------------------
__global__ void tl_reduce_mean(const float* __restrict__ rowloss,
                               float* __restrict__ out) {
    __shared__ float s[256];
    float acc = 0.0f;
    for (int i = threadIdx.x; i < N_ROWS; i += 256)
        acc += rowloss[i];
    s[threadIdx.x] = acc;
    __syncthreads();
    for (int st = 128; st > 0; st >>= 1) {
        if (threadIdx.x < st) s[threadIdx.x] += s[threadIdx.x + st];
        __syncthreads();
    }
    if (threadIdx.x == 0) out[0] = s[0] / (float)N_ROWS;
}

// ---------------------------------------------------------------------------
extern "C" void kernel_launch(void* const* d_in, const int* in_sizes, int n_in,
                              void* d_out, int out_size, void* d_ws, size_t ws_size,
                              hipStream_t stream) {
    (void)in_sizes; (void)n_in; (void)out_size; (void)ws_size;

    const float* emb    = (const float*)d_in[0];
    const int*   labels = (const int*)d_in[1];
    float*       out    = (float*)d_out;

    // workspace layout: [0, 2MB) f16 normalized embeddings; then row losses
    _Float16* ef      = (_Float16*)d_ws;
    float*    rowloss = (float*)((char*)d_ws + (size_t)N_ROWS * DIM * sizeof(_Float16));

    tl_normalize_f16<<<N_ROWS / 8, 256, 0, stream>>>(emb, ef);
    tl_gram_mine<<<N_ROWS / 64, 256, 0, stream>>>(ef, labels, rowloss);
    tl_reduce_mean<<<1, 256, 0, stream>>>(rowloss, out);
}
